// Q_ResNet20_Daq_8074538516493
// MI455X (gfx1250) — 8-bit matrix operand order repaired and checked by reading, compiled, not run on hardware
//
#include <hip/hip_runtime.h>
#include <math.h>

#define QMAXF 127.0f

typedef __attribute__((ext_vector_type(8))) int v8i;
union V8 { int i[8]; v8i v; };


__global__ void zero_scalars_kernel(unsigned* p) {
  if (threadIdx.x < 64) p[threadIdx.x] = 0u;
}

__global__ void absmax_kernel(const float* __restrict__ x, int n,
                              unsigned* __restrict__ out) {
  __shared__ float red[256];
  float m = 0.0f;
  for (int i = blockIdx.x * blockDim.x + threadIdx.x; i < n;
       i += gridDim.x * blockDim.x)
    m = fmaxf(m, fabsf(x[i]));
  red[threadIdx.x] = m;
  __syncthreads();
  for (int st = 128; st > 0; st >>= 1) {
    if ((int)threadIdx.x < st)
      red[threadIdx.x] = fmaxf(red[threadIdx.x], red[threadIdx.x + st]);
    __syncthreads();
  }
  if (threadIdx.x == 0) atomicMax(out, __float_as_uint(red[0]));
}

__global__ void quant_act_nhwc_kernel(const float* __restrict__ x, int Nimg,
                                      int C, int H, int W,
                                      const unsigned* __restrict__ maxbits,
                                      signed char* __restrict__ q) {
  int n = Nimg * C * H * W;
  int i = blockIdx.x * blockDim.x + threadIdx.x;
  if (i >= n) return;
  int c = i % C;
  int t = i / C;
  int w = t % W; t /= W;
  int h = t % H;
  int ni = t / H;
  float s = fmaxf(__uint_as_float(*maxbits), 1e-8f) * (1.0f / QMAXF);
  float v = rintf(x[(((size_t)ni * C + c) * H + h) * W + w] / s);
  v = fminf(fmaxf(v, -QMAXF), QMAXF);
  q[i] = (signed char)(int)v;
}

__global__ void qdq_relu_kernel(float* __restrict__ x, int n,
                                const unsigned* __restrict__ maxbits) {
  int i = blockIdx.x * blockDim.x + threadIdx.x;
  if (i >= n) return;
  float s = fmaxf(__uint_as_float(*maxbits), 1e-8f) * (1.0f / QMAXF);
  float v = rintf(x[i] / s);
  v = fminf(fmaxf(v, -QMAXF), QMAXF) * s;
  x[i] = fmaxf(v, 0.0f);
}

__global__ void weight_quant_kernel(const float* __restrict__ w, int K,
                                    int Cin, int k2,
                                    float* __restrict__ s_w,
                                    signed char* __restrict__ w8) {
  __shared__ float red[256];
  int o = blockIdx.x;
  const float* row = w + (size_t)o * K;
  float m = 0.0f;
  for (int i = threadIdx.x; i < K; i += 256) m = fmaxf(m, fabsf(row[i]));
  red[threadIdx.x] = m;
  __syncthreads();
  for (int st = 128; st > 0; st >>= 1) {
    if ((int)threadIdx.x < st)
      red[threadIdx.x] = fmaxf(red[threadIdx.x], red[threadIdx.x + st]);
    __syncthreads();
  }
  float sc = fmaxf(red[0], 1e-8f) * (1.0f / QMAXF);
  if (threadIdx.x == 0) s_w[o] = sc;
  for (int i = threadIdx.x; i < K; i += 256) {
    float q = rintf(row[i] / sc);
    q = fminf(fmaxf(q, -QMAXF), QMAXF);
    int cin = i / k2;
    int t = i - cin * k2;
    w8[(size_t)o * K + t * Cin + cin] = (signed char)(int)q;
  }
}

__global__ void pack_b_kernel(const signed char* __restrict__ w8, int O, int K,
                              int KC, int NT, unsigned* __restrict__ packB) {
  int idx = blockIdx.x * blockDim.x + threadIdx.x;
  int total = NT * KC * 256;
  if (idx >= total) return;
  int l = idx & 31;
  int v = (idx >> 5) & 7;
  int rest = idx >> 8;
  int kc = rest % KC;
  int nt = rest / KC;
  int o = nt * 16 + (l & 15);
  int kbase = kc * 64 + ((v >> 2) << 5) + ((l >> 4) << 4) + ((v & 3) << 2);
  unsigned d = 0;
  for (int b = 0; b < 4; ++b) {
    int kk = kbase + b;
    unsigned byte = 0;
    if (o < O && kk < K) byte = (unsigned char)w8[(size_t)o * K + kk];
    d |= (byte & 0xFFu) << (b * 8);
  }
  packB[idx] = d;
}

__device__ __forceinline__ int a_koff(int v, int g) {
  return ((v >> 2) << 5) + (g << 4) + ((v & 3) << 2);
}

template <int KSZ, int MT, int NTW>
__global__ __launch_bounds__(256)
void conv_wmma_fast_kernel(const signed char* __restrict__ x8,
                           const unsigned* __restrict__ packB,
                           const float* __restrict__ bias,
                           const float* __restrict__ s_w,
                           const unsigned* __restrict__ act_maxbits,
                           const float* __restrict__ residual,
                           float* __restrict__ out,
                           int Nimg, int cinShift, int H, int W,
                           int O, int Ho, int Wo, int stride, int pad,
                           int relu) {
  const int M = Nimg * Ho * Wo;
  const int wv = threadIdx.x >> 5;
  const int mtile0 = (blockIdx.x * 8 + wv) * MT;
  if (mtile0 * 16 >= M) return;
  const int lane = threadIdx.x & 31;
  const int g = lane >> 4;
  const int hl = lane & 15;
  const int Cin = 1 << cinShift;
  const int K = Cin * KSZ * KSZ;
  const int KC = (K + 63) >> 6;

  int ihb[MT], iwb[MT], ni[MT];
  bool mvalid[MT];
#pragma unroll
  for (int mt = 0; mt < MT; ++mt) {
    int m = (mtile0 + mt) * 16 + hl;
    mvalid[mt] = (m < M);
    int ow = 0, oh = 0, n2 = 0;
    if (mvalid[mt]) { ow = m % Wo; int t = m / Wo; oh = t % Ho; n2 = t / Ho; }
    ihb[mt] = oh * stride - pad;
    iwb[mt] = ow * stride - pad;
    ni[mt] = n2;
  }

  const unsigned* bb[NTW];
#pragma unroll
  for (int nt = 0; nt < NTW; ++nt)
    bb[nt] = packB +
             (((size_t)(blockIdx.y * NTW + nt) * KC) << 8) + lane;

  V8 acc[MT][NTW];
#pragma unroll
  for (int mt = 0; mt < MT; ++mt)
#pragma unroll
    for (int nt = 0; nt < NTW; ++nt)
#pragma unroll
      for (int r = 0; r < 8; ++r) acc[mt][nt].i[r] = 0;

  for (int kc = 0; kc < KC; ++kc) {
    V8 A[MT], B[NTW];
#pragma unroll
    for (int nt = 0; nt < NTW; ++nt) {
      __builtin_prefetch(bb[nt] + (((size_t)kc + 1) << 8), 0, 1);
#pragma unroll
      for (int v = 0; v < 8; ++v)
        B[nt].i[v] = (int)bb[nt][((size_t)kc << 8) + (v << 5)];
    }
#pragma unroll
    for (int v = 0; v < 8; ++v) {
      int kbg = kc * 64 + a_koff(v, g);
      int t = kbg >> cinShift;
      int cb = kbg & (Cin - 1);
      int kh = t / KSZ;
      int kw = t - kh * KSZ;
      bool kok = (kbg < K);
#pragma unroll
      for (int mt = 0; mt < MT; ++mt) {
        int ih = ihb[mt] + kh;
        int iw = iwb[mt] + kw;
        unsigned d = 0;
        if (mvalid[mt] && kok && ih >= 0 && ih < H && iw >= 0 && iw < W)
          d = *(const unsigned*)(x8 +
                (((size_t)ni[mt] * H + ih) * W + iw) * Cin + cb);
        A[mt].i[v] = (int)d;
      }
    }
#pragma unroll
    for (int mt = 0; mt < MT; ++mt)
#pragma unroll
      for (int nt = 0; nt < NTW; ++nt)
        acc[mt][nt].v = __builtin_amdgcn_wmma_i32_16x16x64_iu8(
            true, A[mt].v, true, B[nt].v, acc[mt][nt].v, false, false);
  }

  float s_a = fmaxf(__uint_as_float(*act_maxbits), 1e-8f) * (1.0f / QMAXF);
#pragma unroll
  for (int nt = 0; nt < NTW; ++nt) {
    int col = (blockIdx.y * NTW + nt) * 16 + hl;
    if (col < O) {
      float sb = s_a * s_w[col];
      float bq = rintf(bias[col] / sb);
#pragma unroll
      for (int mt = 0; mt < MT; ++mt) {
#pragma unroll
        for (int r = 0; r < 8; ++r) {
          int row = (mtile0 + mt) * 16 + r + (g << 3);
          if (row < M) {
            int ow2 = row % Wo; int t = row / Wo;
            int oh2 = t % Ho; int n2 = t / Ho;
            size_t oidx = (((size_t)n2 * O + col) * Ho + oh2) * Wo + ow2;
            float val = ((float)acc[mt][nt].i[r] + bq) * sb;
            if (residual) val += residual[oidx];
            if (relu) val = fmaxf(val, 0.0f);
            out[oidx] = val;
          }
        }
      }
    }
  }
}

__global__ __launch_bounds__(256)
void conv_wmma_generic_kernel(const signed char* __restrict__ x8,
                              const unsigned* __restrict__ packB,
                              const float* __restrict__ bias,
                              const float* __restrict__ s_w,
                              const unsigned* __restrict__ act_maxbits,
                              const float* __restrict__ residual,
                              float* __restrict__ out,
                              int Nimg, int Cin, int H, int W,
                              int O, int Ho, int Wo,
                              int ksz, int stride, int pad, int relu) {
  const int M = Nimg * Ho * Wo;
  const int wv = threadIdx.x >> 5;
  const int mtile = blockIdx.x * 8 + wv;
  if (mtile * 16 >= M) return;
  const int lane = threadIdx.x & 31;
  const int g = lane >> 4;
  const int hl = lane & 15;
  const int K = Cin * ksz * ksz;
  const int KC = (K + 63) >> 6;

  int m = mtile * 16 + hl;
  bool mvalid = (m < M);
  int ow = 0, oh = 0, ni = 0;
  if (mvalid) { ow = m % Wo; int t = m / Wo; oh = t % Ho; ni = t / Ho; }
  const int ihb = oh * stride - pad;
  const int iwb = ow * stride - pad;

  const unsigned* bbase = packB + (((size_t)blockIdx.y * KC) << 8) + lane;

  V8 acc;
#pragma unroll
  for (int r = 0; r < 8; ++r) acc.i[r] = 0;

  for (int kc = 0; kc < KC; ++kc) {
    V8 A, B;
#pragma unroll
    for (int v = 0; v < 8; ++v)
      B.i[v] = (int)bbase[((size_t)kc << 8) + (v << 5)];
#pragma unroll
    for (int v = 0; v < 8; ++v) {
      int kb = kc * 64 + a_koff(v, g);
      unsigned d = 0;
      if (mvalid) {
#pragma unroll
        for (int bb = 0; bb < 4; ++bb) {
          int kk = kb + bb;
          unsigned byte = 0;
          if (kk < K) {
            int t = kk / Cin;
            int cin = kk - t * Cin;
            int kh = t / ksz;
            int kw = t - kh * ksz;
            int ih = ihb + kh;
            int iw = iwb + kw;
            if (ih >= 0 && ih < H && iw >= 0 && iw < W)
              byte = (unsigned char)
                  x8[(((size_t)ni * H + ih) * W + iw) * Cin + cin];
          }
          d |= (byte & 0xFFu) << (bb * 8);
        }
      }
      A.i[v] = (int)d;
    }
    acc.v = __builtin_amdgcn_wmma_i32_16x16x64_iu8(true, A.v, true, B.v,
                                                   acc.v, false, false);
  }

  float s_a = fmaxf(__uint_as_float(*act_maxbits), 1e-8f) * (1.0f / QMAXF);
  int col = blockIdx.y * 16 + hl;
  if (col < O) {
    float sb = s_a * s_w[col];
    float bq = rintf(bias[col] / sb);
#pragma unroll
    for (int r = 0; r < 8; ++r) {
      int row = mtile * 16 + r + (g << 3);
      if (row < M) {
        int ow2 = row % Wo; int t = row / Wo; int oh2 = t % Ho; int n2 = t / Ho;
        size_t oidx = (((size_t)n2 * O + col) * Ho + oh2) * Wo + ow2;
        float val = ((float)acc.i[r] + bq) * sb;
        if (residual) val += residual[oidx];
        if (relu) val = fmaxf(val, 0.0f);
        out[oidx] = val;
      }
    }
  }
}

__global__ void avgpool_kernel(const float* __restrict__ x,
                               float* __restrict__ out, int NC, int HW) {
  int i = blockIdx.x * blockDim.x + threadIdx.x;
  if (i >= NC) return;
  const float* p = x + (size_t)i * HW;
  float s = 0.0f;
  for (int j = 0; j < HW; ++j) s += p[j];
  out[i] = s / (float)HW;
}


extern "C" void kernel_launch(void* const* d_in, const int* in_sizes, int n_in,
                              void* d_out, int out_size, void* d_ws,
                              size_t ws_size, hipStream_t stream) {
  (void)in_sizes; (void)n_in; (void)out_size; (void)ws_size;
  const int NB = 1024;

  char* ws = (char*)d_ws;
  unsigned* scal = (unsigned*)ws;
  size_t off = 1024;
  float* bufA = (float*)(ws + off); off += (size_t)64 << 20;
  float* bufB = (float*)(ws + off); off += (size_t)64 << 20;
  float* bufC = (float*)(ws + off); off += (size_t)32 << 20;
  signed char* X8 = (signed char*)(ws + off); off += (size_t)16 << 20;
  float* pooled = (float*)(ws + off); off += (size_t)NB * 64 * 4;
  float* s_wbuf = (float*)(ws + off); off += 512;
  signed char* w8 = (signed char*)(ws + off); off += 64 * 1024;
  unsigned* packB = (unsigned*)(ws + off); off += 64 * 1024;

  zero_scalars_kernel<<<1, 64, 0, stream>>>(scal);

  auto absmax = [&](const float* p, int n, int slot) {
    int blocks = (n + 255) / 256; if (blocks > 1024) blocks = 1024;
    absmax_kernel<<<blocks, 256, 0, stream>>>(p, n, scal + slot);
  };
  auto quantnhwc = [&](const float* p, int C, int H, int W, int slot) {
    int n = NB * C * H * W;
    quant_act_nhwc_kernel<<<(n + 255) / 256, 256, 0, stream>>>(p, NB, C, H, W,
                                                               scal + slot, X8);
  };
  auto qdqrelu = [&](float* p, int n, int slot) {
    qdq_relu_kernel<<<(n + 255) / 256, 256, 0, stream>>>(p, n, scal + slot);
  };
  auto runconv = [&](const signed char* xq, const float* w, const float* b,
                     int O, int Cin, int k, int slot, int H, int W, int stride,
                     int pad, const float* res, float* outp, int relu) {
    int K = Cin * k * k;
    int KC = (K + 63) / 64;
    int NT = (O + 15) / 16;
    int sh = 0; while ((1 << sh) < Cin) ++sh;
    bool pow2 = ((1 << sh) == Cin) && (Cin >= 4);
    int NTW = (pow2 && NT >= 2) ? 2 : 1;
    int NTpad = ((NT + NTW - 1) / NTW) * NTW;

    weight_quant_kernel<<<O, 256, 0, stream>>>(w, K, Cin, k * k, s_wbuf, w8);
    int total = NTpad * KC * 256;
    pack_b_kernel<<<(total + 255) / 256, 256, 0, stream>>>(w8, O, K, KC, NTpad,
                                                           packB);
    int Ho = (H + 2 * pad - k) / stride + 1;
    int Wo = (W + 2 * pad - k) / stride + 1;
    int M = NB * Ho * Wo;
    int mtiles = (M + 15) / 16;
    if (pow2) {
      dim3 grid((mtiles + 15) / 16, NTpad / NTW);
      if (k == 3 && NTW == 2)
        conv_wmma_fast_kernel<3, 2, 2><<<grid, 256, 0, stream>>>(
            xq, packB, b, s_wbuf, scal + slot, res, outp, NB, sh, H, W, O, Ho,
            Wo, stride, pad, relu);
      else if (k == 3)
        conv_wmma_fast_kernel<3, 2, 1><<<grid, 256, 0, stream>>>(
            xq, packB, b, s_wbuf, scal + slot, res, outp, NB, sh, H, W, O, Ho,
            Wo, stride, pad, relu);
      else if (NTW == 2)
        conv_wmma_fast_kernel<1, 2, 2><<<grid, 256, 0, stream>>>(
            xq, packB, b, s_wbuf, scal + slot, res, outp, NB, sh, H, W, O, Ho,
            Wo, stride, pad, relu);
      else
        conv_wmma_fast_kernel<1, 2, 1><<<grid, 256, 0, stream>>>(
            xq, packB, b, s_wbuf, scal + slot, res, outp, NB, sh, H, W, O, Ho,
            Wo, stride, pad, relu);
    } else {
      dim3 grid((mtiles + 7) / 8, NTpad);
      conv_wmma_generic_kernel<<<grid, 256, 0, stream>>>(
          xq, packB, b, s_wbuf, scal + slot, res, outp, NB, Cin, H, W, O, Ho,
          Wo, k, stride, pad, relu);
    }
  };

  const float* x = (const float*)d_in[0];
  int ip = 1;
  auto P = [&]() { return (const float*)d_in[ip++]; };
  const float* init_w = P(); const float* init_b = P();
  const float *w1[9], *b1[9], *w2[9], *b2[9], *wid[9], *bid[9];
  int outc_[9], resize_[9];
  {
    int stage_ch[3] = {16, 32, 64};
    for (int i = 0; i < 9; ++i) {
      int s = i / 3, u = i % 3;
      int rs = (u == 0 && s > 0) ? 1 : 0;
      w1[i] = P(); b1[i] = P(); w2[i] = P(); b2[i] = P();
      if (rs) { wid[i] = P(); bid[i] = P(); } else { wid[i] = 0; bid[i] = 0; }
      outc_[i] = stage_ch[s]; resize_[i] = rs;
    }
  }
  const float* fc_w = P(); const float* fc_b = P();

  int slot = 0;
  absmax(x, NB * 3 * 32 * 32, slot);
  quantnhwc(x, 3, 32, 32, slot);
  runconv(X8, init_w, init_b, 16, 3, 3, slot, 32, 32, 1, 1, nullptr, bufA, 0);
  slot++;
  int H = 32, C = 16;
  absmax(bufA, NB * C * H * H, slot);
  qdqrelu(bufA, NB * C * H * H, slot);
  slot++;

  for (int i = 0; i < 9; ++i) {
    int oc = outc_[i];
    if (!resize_[i]) {
      absmax(bufA, NB * C * H * H, slot);
      quantnhwc(bufA, C, H, H, slot);
      runconv(X8, w1[i], b1[i], oc, C, 3, slot, H, H, 1, 1, nullptr, bufB, 1);
      slot++;
      int nb = NB * oc * H * H;
      absmax(bufB, nb, slot);
      quantnhwc(bufB, oc, H, H, slot);
      runconv(X8, w2[i], b2[i], oc, oc, 3, slot, H, H, 1, 1, bufA, bufA, 0);
      slot++;
      absmax(bufA, nb, slot);
      qdqrelu(bufA, nb, slot);
      slot++;
      C = oc;
    } else {
      absmax(bufA, NB * C * H * H, slot);
      quantnhwc(bufA, C, H, H, slot);
      int Ho = H / 2;
      runconv(X8, wid[i], bid[i], oc, C, 1, slot, H, H, 2, 0, nullptr, bufB, 0);
      runconv(X8, w1[i], b1[i], oc, C, 3, slot, H, H, 2, 1, nullptr, bufC, 1);
      slot++;
      int nb = NB * oc * Ho * Ho;
      absmax(bufC, nb, slot);
      quantnhwc(bufC, oc, Ho, Ho, slot);
      runconv(X8, w2[i], b2[i], oc, oc, 3, slot, Ho, Ho, 1, 1, bufB, bufA, 0);
      slot++;
      absmax(bufA, nb, slot);
      qdqrelu(bufA, nb, slot);
      slot++;
      H = Ho; C = oc;
    }
  }

  avgpool_kernel<<<(NB * 64 + 255) / 256, 256, 0, stream>>>(bufA, pooled,
                                                            NB * 64, 64);
  absmax(pooled, NB * 64, slot);
  quantnhwc(pooled, 64, 1, 1, slot);
  runconv(X8, fc_w, fc_b, 100, 64, 1, slot, 1, 1, 1, 0, nullptr,
          (float*)d_out, 0);
}
